// VariationalLSTM_35862976922290
// MI455X (gfx1250) — compile-verified, matrix operand order checked by reading, not run on hardware
//
#include <hip/hip_runtime.h>
#include <hip/hip_bf16.h>
#include <stdint.h>

// ---------------- problem constants ----------------
#define BB 128
#define TT 512
#define II 512
#define HH 1024
#define GG (4 * HH)      // 4096 gate columns
#define NBLK (HH / 32)   // 32 persistent blocks for the recurrent scan
#define LDSW 40          // padded LDS row stride (halfs): 20 dwords, conflict-free

typedef _Float16 v8h  __attribute__((ext_vector_type(8)));
typedef _Float16 v16h __attribute__((ext_vector_type(16)));
typedef float    v8f  __attribute__((ext_vector_type(8)));

// ---------------- CDNA5 async global->LDS copy (ASYNCcnt tracked) ----------
#if __has_builtin(__builtin_amdgcn_global_load_async_to_lds_b128)
#define HAVE_ASYNC 1
#else
#define HAVE_ASYNC 0
#endif

typedef int v4i_n __attribute__((vector_size(16)));
typedef __attribute__((address_space(1))) v4i_n* as1_v4i;
typedef __attribute__((address_space(3))) v4i_n* as3_v4i;

__device__ __forceinline__ void cp16_async(const _Float16* g, _Float16* l) {
#if HAVE_ASYNC
  __builtin_amdgcn_global_load_async_to_lds_b128(
      (as1_v4i)(uintptr_t)g, (as3_v4i)(unsigned)(uintptr_t)l, 0, 0);
#else
  *(uint4*)l = *(const uint4*)g;
#endif
}

template <int N>
__device__ __forceinline__ void wait_async() {
#if HAVE_ASYNC
#if __has_builtin(__builtin_amdgcn_s_wait_asynccnt)
  __builtin_amdgcn_s_wait_asynccnt(N);
#else
  asm volatile("s_wait_asynccnt %0" ::"n"(N) : "memory");
#endif
#endif
}

// Load one 16x32 (f16) A/B WMMA fragment from an LDS row (row-major, LDSW
// halfs per row). Lane layout per CDNA5 ISA 7.12.2: lanes 0-15 ->
// K {0..7,16..23}, lanes 16-31 -> K {8..15,24..31}.
__device__ __forceinline__ v16h load_frag(const _Float16* row, int lane) {
  const int kh = (lane & 16) ? 8 : 0;
  v8h lo = *(const v8h*)(row + kh);
  v8h hi = *(const v8h*)(row + kh + 16);
  return __builtin_shufflevector(lo, hi, 0, 1, 2, 3, 4, 5, 6, 7,
                                         8, 9, 10, 11, 12, 13, 14, 15);
}

__device__ __forceinline__ float sigmoidf_(float x) {
  return 1.0f / (1.0f + __expf(-x));
}

// ---------------- fp32 -> fp16 convert ----------------
__global__ __launch_bounds__(256) void cvt_f32_f16(const float* __restrict__ in,
                                                   _Float16* __restrict__ out,
                                                   long n) {
  long i = (((long)blockIdx.x * blockDim.x) + threadIdx.x) * 4;
  if (i + 3 < n) {
    float4 v = *(const float4*)(in + i);
    out[i + 0] = (_Float16)v.x;
    out[i + 1] = (_Float16)v.y;
    out[i + 2] = (_Float16)v.z;
    out[i + 3] = (_Float16)v.w;
  }
}

// ---------------- zero h state ----------------
__global__ __launch_bounds__(256) void zero_state(_Float16* __restrict__ h,
                                                  int n) {
  int i = blockIdx.x * blockDim.x + threadIdx.x;
  if (i < n) h[i] = (_Float16)0.0f;
}

__global__ __launch_bounds__(256) void zero_i32(int* __restrict__ p, int n) {
  int i = blockIdx.x * blockDim.x + threadIdx.x;
  if (i < n) p[i] = 0;
}

// ---------------- parallel input-projection GEMM ----------------
// xg[m, g] = sum_k A[m, k] * W[g, k] + bi[g] + bh[g], m = t*B + b, fp16 out.
// Block tile 128M x 64N, 8 waves (wave = m-tile), double-buffered async LDS.
__global__ __launch_bounds__(256) void gemm_xg(
    const _Float16* __restrict__ Abase, int strideT, int strideB,
    const _Float16* __restrict__ W, int K,
    const float* __restrict__ bi, const float* __restrict__ bh,
    _Float16* __restrict__ xg) {
  __shared__ _Float16 As[2][128 * LDSW];  // 2 x 10 KB
  __shared__ _Float16 Bs[2][64 * LDSW];   // 2 x 5 KB

  const int tid = threadIdx.x;
  const int lane = tid & 31;
  const int wave = tid >> 5;
  const int m0 = blockIdx.x * 128;
  const int n0 = blockIdx.y * 64;

  v8f acc[4];
#pragma unroll
  for (int j = 0; j < 4; ++j) {
    int g = n0 + j * 16 + (lane & 15);
    float bv = bi[g] + bh[g];
#pragma unroll
    for (int e = 0; e < 8; ++e) acc[j][e] = bv;
  }

  const int r = tid & 127;    // A row in tile
  const int half = tid >> 7;  // which 16-half chunk of the A row
  const int m = m0 + r;
  const _Float16* arow =
      Abase + (size_t)(m >> 7) * strideT + (size_t)(m & 127) * strideB;
  const int browB = tid >> 2;  // Bs row (64 rows, 4 threads each)
  const int bqtr = tid & 3;    // 8-half chunk within Bs row
  const size_t woff = (size_t)(n0 + browB) * K;

  auto stage = [&](int buf, int k0) {
    const _Float16* a = arow + k0 + half * 16;
    _Float16* al = &As[buf][r * LDSW + half * 16];
    cp16_async(a, al);
    cp16_async(a + 8, al + 8);
    cp16_async(W + woff + k0 + bqtr * 8, &Bs[buf][browB * LDSW + bqtr * 8]);
  };

  const int NK = K >> 5;
  stage(0, 0);
#pragma unroll 2
  for (int kc = 0; kc < NK; ++kc) {
    const int cur = kc & 1;
    if (kc + 1 < NK) {
      stage(cur ^ 1, (kc + 1) * 32);
      wait_async<3>();  // drain prior batch (3 asyncs/wave), new batch in flight
    } else {
      wait_async<0>();
    }
    __syncthreads();
    // one clause of fragment loads, then a back-to-back WMMA chain
    v16h af = load_frag(&As[cur][((wave << 4) + (lane & 15)) * LDSW], lane);
    v16h bf[4];
#pragma unroll
    for (int j = 0; j < 4; ++j)
      bf[j] = load_frag(&Bs[cur][(j * 16 + (lane & 15)) * LDSW], lane);
#pragma unroll
    for (int j = 0; j < 4; ++j)
      acc[j] = __builtin_amdgcn_wmma_f32_16x16x32_f16(
          false, af, false, bf[j], (short)0, acc[j], false, false);
    __syncthreads();
  }

  // D layout: VGPR e -> M = e (lanes 0-15) / e+8 (lanes 16-31); N = lane&15.
#pragma unroll
  for (int j = 0; j < 4; ++j) {
    int g = n0 + j * 16 + (lane & 15);
#pragma unroll
    for (int e = 0; e < 8; ++e) {
      int mm = m0 + (wave << 4) + e + ((lane & 16) ? 8 : 0);
      xg[(size_t)mm * GG + g] = (_Float16)acc[j][e];
    }
  }
}

// ---------------- persistent recurrent scan ----------------
// 32 resident blocks; block owns 32 hidden units (all 4 gates) x 128 batch.
// Loops t = 0..T-1 internally; inter-block sync via atomic arrive counters.
// c state and variational masks live entirely in registers (block-private).
__global__ __launch_bounds__(256) void lstm_layer(
    const _Float16* __restrict__ Whh, const _Float16* __restrict__ xg,
    _Float16* __restrict__ hA, _Float16* __restrict__ hB,
    const float* __restrict__ om, const float* __restrict__ hmask,
    const float* __restrict__ cmask,
    _Float16* __restrict__ ys, float* __restrict__ dout,
    float* __restrict__ hf, float* __restrict__ cf, int mode,
    int* __restrict__ syncc) {
  __shared__ _Float16 As[2][128 * LDSW];  // h tile (2 x 10 KB)
  __shared__ _Float16 Bs[2][128 * LDSW];  // W tile (4q x 32u rows, 2 x 10 KB)

  const int tid = threadIdx.x;
  const int lane = tid & 31;
  const int wave = tid >> 5;
  const int h0 = blockIdx.x * 32;
  const int r = tid & 127;
  const int half = tid >> 7;

  // This block's W_hh staging row (fixed across all timesteps; L2 resident).
  const _Float16* wrow = Whh + (size_t)((r >> 5) * HH + h0 + (r & 31)) * HH;

  // hoist loop-invariant masks and keep c entirely in registers
  float omr[2][8], hmr[2][8], cmr[2][8], creg[2][8];
#pragma unroll
  for (int j = 0; j < 2; ++j) {
    int u = h0 + (j << 4) + (lane & 15);
#pragma unroll
    for (int e = 0; e < 8; ++e) {
      int b = (wave << 4) + e + ((lane & 16) ? 8 : 0);
      size_t m = (size_t)b * HH + u;
      omr[j][e] = om[m];
      hmr[j][e] = hmask[m];
      cmr[j][e] = cmask[m];
      creg[j][e] = 0.0f;
    }
  }

#pragma unroll 1
  for (int t = 0; t < TT; ++t) {
    const _Float16* hp = (t & 1) ? hB : hA;
    _Float16* hn = (t & 1) ? hA : hB;

    auto stage = [&](int buf, int k0) {
      const _Float16* a = hp + (size_t)r * HH + k0 + half * 16;
      _Float16* al = &As[buf][r * LDSW + half * 16];
      cp16_async(a, al);
      cp16_async(a + 8, al + 8);
      const _Float16* b = wrow + k0 + half * 16;
      _Float16* bl = &Bs[buf][r * LDSW + half * 16];
      cp16_async(b, bl);
      cp16_async(b + 8, bl + 8);
    };

    stage(0, 0);

    // prefetch xg pre-activations; latency hides under the whole K loop
    float xv[4][2][8];
#pragma unroll
    for (int q = 0; q < 4; ++q)
#pragma unroll
      for (int j = 0; j < 2; ++j) {
        int u = h0 + (j << 4) + (lane & 15);
        int g = q * HH + u;
#pragma unroll
        for (int e = 0; e < 8; ++e) {
          int b = (wave << 4) + e + ((lane & 16) ? 8 : 0);
          xv[q][j][e] = (float)xg[((size_t)t * BB + b) * GG + g];
        }
      }

    v8f acc[4][2];
#pragma unroll
    for (int q = 0; q < 4; ++q)
#pragma unroll
      for (int j = 0; j < 2; ++j)
#pragma unroll
        for (int e = 0; e < 8; ++e) acc[q][j][e] = 0.0f;

#pragma unroll 2
    for (int kc = 0; kc < HH / 32; ++kc) {
      const int cur = kc & 1;
      if (kc + 1 < HH / 32) {
        stage(cur ^ 1, (kc + 1) * 32);
        wait_async<4>();  // drain prior batch (4 asyncs/wave)
      } else {
        wait_async<0>();
      }
      __syncthreads();
      v16h af = load_frag(&As[cur][((wave << 4) + (lane & 15)) * LDSW], lane);
      v16h bf[4][2];
#pragma unroll
      for (int q = 0; q < 4; ++q)
#pragma unroll
        for (int j = 0; j < 2; ++j)
          bf[q][j] = load_frag(
              &Bs[cur][((q << 5) + (j << 4) + (lane & 15)) * LDSW], lane);
#pragma unroll
      for (int q = 0; q < 4; ++q)
#pragma unroll
        for (int j = 0; j < 2; ++j)
          acc[q][j] = __builtin_amdgcn_wmma_f32_16x16x32_f16(
              false, af, false, bf[q][j], (short)0, acc[q][j], false, false);
      __syncthreads();
    }

    // fused LSTM cell + variational masks (c state in registers)
#pragma unroll
    for (int j = 0; j < 2; ++j) {
      int u = h0 + (j << 4) + (lane & 15);
#pragma unroll
      for (int e = 0; e < 8; ++e) {
        int b = (wave << 4) + e + ((lane & 16) ? 8 : 0);
        size_t m = (size_t)b * HH + u;
        float iv = sigmoidf_(acc[0][j][e] + xv[0][j][e]);
        float fv = sigmoidf_(acc[1][j][e] + xv[1][j][e]);
        float gv = tanhf(acc[2][j][e] + xv[2][j][e]);
        float ov = sigmoidf_(acc[3][j][e] + xv[3][j][e]);
        float c2 = fv * creg[j][e] + iv * gv;
        float h2 = ov * tanhf(c2);
        float hnv = h2 * hmr[j][e];
        float cnv = c2 * cmr[j][e];
        float outv = h2 * omr[j][e];
        creg[j][e] = cnv;
        hn[m] = (_Float16)hnv;
        if (mode == 0) {
          ys[(size_t)t * (BB * HH) + m] = (_Float16)outv;
        } else {
          dout[(size_t)b * (TT * HH) + (size_t)t * HH + u] = outv;
          if (t == TT - 1) {
            hf[m] = hnv;
            cf[m] = cnv;
          }
        }
      }
    }

    // grid-wide step barrier (release h writes, acquire for next step)
    if (t + 1 < TT) {
      __threadfence();  // each wave drains its own h stores to device scope
      __syncthreads();
      if (tid == 0) {
        atomicAdd(&syncc[t], 1);
        while (__hip_atomic_load(&syncc[t], __ATOMIC_RELAXED,
                                 __HIP_MEMORY_SCOPE_AGENT) < NBLK)
          __builtin_amdgcn_s_sleep(1);
      }
      __syncthreads();
      __threadfence();  // acquire before reading other blocks' h
    }
  }
}

// ---------------- host-side orchestration ----------------
extern "C" void kernel_launch(void* const* d_in, const int* in_sizes, int n_in,
                              void* d_out, int out_size, void* d_ws,
                              size_t ws_size, hipStream_t stream) {
  (void)in_sizes; (void)n_in; (void)out_size; (void)ws_size;

  const float* x    = (const float*)d_in[0];
  const float* Wih0 = (const float*)d_in[1];
  const float* Whh0 = (const float*)d_in[2];
  const float* bih0 = (const float*)d_in[3];
  const float* bhh0 = (const float*)d_in[4];
  const float* Wih1 = (const float*)d_in[5];
  const float* Whh1 = (const float*)d_in[6];
  const float* bih1 = (const float*)d_in[7];
  const float* bhh1 = (const float*)d_in[8];
  const float* om   = (const float*)d_in[9];    // [L,B,H]
  const float* hm   = (const float*)d_in[10];   // [L,B,H]
  const float* cm   = (const float*)d_in[11];   // [L,B,H]
  float* dout = (float*)d_out;

  char* w = (char*)d_ws;
  auto take = [&](size_t bytes) {
    char* p = w;
    w += (bytes + 255) & ~(size_t)255;
    return p;
  };
  _Float16* xh    = (_Float16*)take((size_t)BB * TT * II * 2);   // 64 MB
  _Float16* Wih0h = (_Float16*)take((size_t)GG * II * 2);        // 4 MB
  _Float16* Whh0h = (_Float16*)take((size_t)GG * HH * 2);        // 8 MB
  _Float16* Wih1h = (_Float16*)take((size_t)GG * HH * 2);        // 8 MB
  _Float16* Whh1h = (_Float16*)take((size_t)GG * HH * 2);        // 8 MB
  _Float16* ys0   = (_Float16*)take((size_t)TT * BB * HH * 2);   // 128 MB
  _Float16* xg    = (_Float16*)take((size_t)TT * BB * GG * 2);   // 512 MB
  _Float16* hA    = (_Float16*)take((size_t)BB * HH * 2);
  _Float16* hB    = (_Float16*)take((size_t)BB * HH * 2);
  int*      syncc = (int*)take((size_t)TT * 4);

  auto cvt = [&](const float* in, _Float16* outp, long n) {
    cvt_f32_f16<<<dim3((unsigned)(n / 4 / 256)), 256, 0, stream>>>(in, outp, n);
  };
  cvt(x, xh, (long)BB * TT * II);
  cvt(Wih0, Wih0h, (long)GG * II);
  cvt(Whh0, Whh0h, (long)GG * HH);
  cvt(Wih1, Wih1h, (long)GG * HH);
  cvt(Whh1, Whh1h, (long)GG * HH);

  float* hf = dout + (size_t)BB * TT * HH;
  float* cf = hf + (size_t)BB * HH;

  const dim3 ggrid(TT * BB / 128, GG / 64);  // (512, 64)

  for (int l = 0; l < 2; ++l) {
    // parallel input projection for every timestep: xg = A * W_ih^T + b
    if (l == 0) {
      gemm_xg<<<ggrid, 256, 0, stream>>>(xh, II, TT * II, Wih0h, II,
                                         bih0, bhh0, xg);
    } else {
      gemm_xg<<<ggrid, 256, 0, stream>>>(ys0, BB * HH, HH, Wih1h, HH,
                                         bih1, bhh1, xg);
    }
    zero_state<<<(BB * HH) / 256, 256, 0, stream>>>(hA, BB * HH);
    zero_i32<<<2, 256, 0, stream>>>(syncc, TT);

    const _Float16* Whh = (l == 0) ? Whh0h : Whh1h;
    lstm_layer<<<NBLK, 256, 0, stream>>>(
        Whh, xg, hA, hB, om + (size_t)l * BB * HH, hm + (size_t)l * BB * HH,
        cm + (size_t)l * BB * HH, ys0, dout, hf, cf, l, syncc);
  }
}
